// Decoder_1271310319982
// MI455X (gfx1250) — matrix operand order repaired and checked by reading, not compiled, not run on hardware
//
#include <hip/hip_runtime.h>
#include <hip/hip_bf16.h>
#include <cstdint>
#include <cstddef>

// ---------------------------------------------------------------------------
// MI455X (gfx1250) transformer decoder forward, wave32 + WMMA bf16.
//  * Weight GEMMs: A[M][K](bf16, double-buffered LDS, ds_load_b128 fragments,
//    buffer index compile-time via 2x manual unroll)
//    x Bt[N][K](bf16, fragment-direct global_load_b128 + global_prefetch).
//  * Attention: flash-style, 1 wave / 16 query rows, QK^T and P*V on WMMA.
// ---------------------------------------------------------------------------

typedef __bf16 bf16_t;
typedef __attribute__((ext_vector_type(16))) __bf16 v16bf;
typedef __attribute__((ext_vector_type(8)))  __bf16 v8bf;
typedef __attribute__((ext_vector_type(8)))  float  v8f;

static constexpr int V_SIZE  = 32000;
static constexpr int D_MODEL = 1024;
static constexpr int DEPTH_L = 4;
static constexpr int NHEAD   = 16;
static constexpr int DHEAD   = 64;
static constexpr int FFDIM   = 4096;
static constexpr int SEQ     = 2048;
static constexpr int QKVDIM  = 3 * NHEAD * DHEAD;   // 3072

#define WMMA_BF16(A_, B_, C_) \
  __builtin_amdgcn_wmma_f32_16x16x32_bf16(false, (A_), false, (B_), (short)0, (C_), false, false)

#define CAT16(lo_, hi_) \
  __builtin_shufflevector((lo_), (hi_), 0, 1, 2, 3, 4, 5, 6, 7, 8, 9, 10, 11, 12, 13, 14, 15)

__device__ __forceinline__ bf16_t f2bf(float f) {
  unsigned u = __float_as_uint(f);
  unsigned r = (u + 0x7FFFu + ((u >> 16) & 1u)) >> 16;   // RNE
  unsigned short s = (unsigned short)r;
  return __builtin_bit_cast(bf16_t, s);
}

// ---------------- weight transpose + f32->bf16:  [K][N] -> [N][K] ----------
__global__ __launch_bounds__(256)
void transpose_f32_bf16(const float* __restrict__ in, bf16_t* __restrict__ out,
                        int K, int N) {
  __shared__ float tile[32][33];
  const int kb = blockIdx.y * 32, nb = blockIdx.x * 32;
  const int tx = threadIdx.x, ty = threadIdx.y;   // block (32,8)
#pragma unroll
  for (int i = ty; i < 32; i += 8)
    tile[i][tx] = in[(size_t)(kb + i) * N + nb + tx];
  __syncthreads();
#pragma unroll
  for (int i = ty; i < 32; i += 8)
    out[(size_t)(nb + i) * K + kb + tx] = f2bf(tile[tx][i]);
}

// ------------------------------ embedding ----------------------------------
__global__ void embed_kernel(const int* __restrict__ x,
                             const float* __restrict__ emb,
                             float* __restrict__ h) {
  int n = blockIdx.x;
  int row = x[n];
  const float4* src = (const float4*)(emb + (size_t)row * D_MODEL);
  float4* dst = (float4*)(h + (size_t)n * D_MODEL);
  for (int d = threadIdx.x; d < D_MODEL / 4; d += blockDim.x) dst[d] = src[d];
}

// ------------------------------ RMSNorm ------------------------------------
__global__ __launch_bounds__(256)
void rmsnorm_kernel(const float* __restrict__ h,
                    const float* __restrict__ gamma,
                    bf16_t* __restrict__ out) {
  __shared__ float red[256];
  int n = blockIdx.x;
  const float* row = h + (size_t)n * D_MODEL;
  float ss = 0.f;
  for (int d = threadIdx.x; d < D_MODEL; d += 256) { float v = row[d]; ss += v * v; }
  red[threadIdx.x] = ss;
  __syncthreads();
  for (int s = 128; s > 0; s >>= 1) {
    if (threadIdx.x < s) red[threadIdx.x] += red[threadIdx.x + s];
    __syncthreads();
  }
  float inv = 32.0f / fmaxf(sqrtf(red[0]), 1e-12f);   // sqrt(1024) = 32
  for (int d = threadIdx.x; d < D_MODEL; d += 256)
    out[(size_t)n * D_MODEL + d] = f2bf(row[d] * inv * gamma[d]);
}

// --------------------------- RoPE + head split -----------------------------
// qkv: [N][3*H*DH] f32 -> Q,K: [H][N][DH] bf16 (RoPE), V: [H][DH][N] bf16
__global__ __launch_bounds__(64)
void rope_split_kernel(const float* __restrict__ qkv,
                       bf16_t* __restrict__ Qb, bf16_t* __restrict__ Kb,
                       bf16_t* __restrict__ Vt) {
  int n = blockIdx.x, hh = blockIdx.y, d = threadIdx.x;
  const float* base = qkv + (size_t)n * QKVDIM;
  const float* qp = base + (0 * NHEAD + hh) * DHEAD;
  const float* kp = base + (1 * NHEAD + hh) * DHEAD;
  const float* vp = base + (2 * NHEAD + hh) * DHEAD;
  float qv = qp[d], kv = kp[d], vv = vp[d];
  int i2 = d & ~1;
  float ang = (float)n * powf(10000.f, -(float)i2 / (float)DHEAD);
  float c = cosf(ang), s = sinf(ang);
  float qr = (d & 1) ? qp[d - 1] : -qp[d + 1];
  float kr = (d & 1) ? kp[d - 1] : -kp[d + 1];
  size_t o = ((size_t)hh * SEQ + n) * DHEAD + d;
  Qb[o] = f2bf(qv * c + qr * s);
  Kb[o] = f2bf(kv * c + kr * s);
  Vt[((size_t)hh * DHEAD + d) * SEQ + n] = f2bf(vv);
}

// ------------------------------- GEMM --------------------------------------
// epilogue( A[M][K](bf16) x Bt[N][K](bf16)^T )
// 256 threads = 8 waves; block tile 128(M) x 128(N) x 32(K).
// Waves: 4(M) x 2(N); each wave 2x4 WMMA tiles -> 8 v8f accumulators.
// A tile double-buffered in LDS with buffer index as a compile-time constant
// (K-loop manually unrolled 2x; all K here are multiples of 64). One barrier
// per 32-wide sub-step; next sub-step's global loads are issued before WMMAs.
// EP: 0 = store f32; 1 = C += acc; 2 = bf16(gelu(acc+bias)); 3 = C += acc+bias

#define GEMM_STEP(BUF_, KB_)                                                  \
  do {                                                                        \
    v16bf a0 = CAT16(*(const v8bf*)&As[BUF_][am0][koff],                      \
                     *(const v8bf*)&As[BUF_][am0][koff + 8]);                 \
    v16bf a1 = CAT16(*(const v8bf*)&As[BUF_][am0 + 16][koff],                 \
                     *(const v8bf*)&As[BUF_][am0 + 16][koff + 8]);            \
    {                                                                         \
      v16bf b = CAT16(*(const v8bf*)(brow0 + (KB_) + khl),                    \
                      *(const v8bf*)(brow0 + (KB_) + khl + 8));               \
      acc[0] = WMMA_BF16(a0, b, acc[0]);                                      \
      acc[1] = WMMA_BF16(a1, b, acc[1]);                                      \
    }                                                                         \
    {                                                                         \
      v16bf b = CAT16(*(const v8bf*)(brow1 + (KB_) + khl),                    \
                      *(const v8bf*)(brow1 + (KB_) + khl + 8));               \
      acc[2] = WMMA_BF16(a0, b, acc[2]);                                      \
      acc[3] = WMMA_BF16(a1, b, acc[3]);                                      \
    }                                                                         \
    {                                                                         \
      v16bf b = CAT16(*(const v8bf*)(brow2 + (KB_) + khl),                    \
                      *(const v8bf*)(brow2 + (KB_) + khl + 8));               \
      acc[4] = WMMA_BF16(a0, b, acc[4]);                                      \
      acc[5] = WMMA_BF16(a1, b, acc[5]);                                      \
    }                                                                         \
    {                                                                         \
      v16bf b = CAT16(*(const v8bf*)(brow3 + (KB_) + khl),                    \
                      *(const v8bf*)(brow3 + (KB_) + khl + 8));               \
      acc[6] = WMMA_BF16(a0, b, acc[6]);                                      \
      acc[7] = WMMA_BF16(a1, b, acc[7]);                                      \
    }                                                                         \
  } while (0)

template <int EP>
__global__ __launch_bounds__(256)
void gemm_bf16(const bf16_t* __restrict__ A, const bf16_t* __restrict__ Bt,
               float* __restrict__ C, bf16_t* __restrict__ Cb,
               const float* __restrict__ bias, int M, int N, int K) {
  __shared__ bf16_t As[2][128][40];   // +8 pad, 16B-aligned rows

  const int tid  = threadIdx.x;
  const int lane = tid & 31;
  const int w    = tid >> 5;
  const int wm   = w & 3, wn = w >> 2;
  const int m0   = blockIdx.y * 128;
  const int n0   = blockIdx.x * 128;

  const int arow = tid >> 1;          // 0..127
  const int acol = (tid & 1) * 16;    // 0,16
  const int r    = lane & 15;
  const int koff = (lane < 16) ? 0 : 16;  // A-frag K offset per half-wave
  const int khl  = (lane < 16) ? 0 : 16;  // B-frag K half per half-wave

  v8f zero = {};
  v8f acc[8];
#pragma unroll
  for (int t = 0; t < 8; ++t) acc[t] = zero;

  const int am0 = wm * 32 + r;
  const bf16_t* arowp = A + (size_t)(m0 + arow) * K + acol;
  // per-lane row pointers into Bt for this wave's 4 column tiles
  const bf16_t* brow0 = Bt + (size_t)(n0 + wn * 64 + 0  + r) * K;
  const bf16_t* brow1 = Bt + (size_t)(n0 + wn * 64 + 16 + r) * K;
  const bf16_t* brow2 = Bt + (size_t)(n0 + wn * 64 + 32 + r) * K;
  const bf16_t* brow3 = Bt + (size_t)(n0 + wn * 64 + 48 + r) * K;

  uint4* const asl0 = (uint4*)(&As[0][arow][acol]);
  uint4* const asl1 = (uint4*)(&As[1][arow][acol]);

  // preload tile 0 into buffer 0
  uint4 pf0 = ((const uint4*)arowp)[0];
  uint4 pf1 = ((const uint4*)arowp)[1];
  asl0[0] = pf0;
  asl0[1] = pf1;
  __syncthreads();

  // K is a multiple of 64: two 32-wide sub-steps per iteration,
  // buffer index is a compile-time constant in each sub-step.
  for (int kb = 0; kb < K; kb += 64) {
    // ---- sub-step A: compute tile kb from As[0], stage tile kb+32 -> As[1]
    {
      const uint4* ag = (const uint4*)(arowp + kb + 32);
      pf0 = ag[0];
      pf1 = ag[1];
    }
    if (kb + 128 < K) {  // warm L2 for the streamed weight rows
      __builtin_prefetch(brow0 + kb + 128, 0, 1);
      __builtin_prefetch(brow1 + kb + 128, 0, 1);
      __builtin_prefetch(brow2 + kb + 128, 0, 1);
      __builtin_prefetch(brow3 + kb + 128, 0, 1);
    }
    GEMM_STEP(0, kb);
    asl1[0] = pf0;
    asl1[1] = pf1;
    __syncthreads();

    // ---- sub-step B: compute tile kb+32 from As[1], stage tile kb+64 -> As[0]
    const bool more = (kb + 64) < K;
    if (more) {
      const uint4* ag = (const uint4*)(arowp + kb + 64);
      pf0 = ag[0];
      pf1 = ag[1];
    }
    GEMM_STEP(1, kb + 32);
    if (more) {
      asl0[0] = pf0;
      asl0[1] = pf1;
    }
    __syncthreads();
  }

  // Epilogue: C layout = VGPR v -> row (v | 8+v), lane&15 -> col
  const int rbase = m0 + wm * 32 + ((lane < 16) ? 0 : 8);
  const int cbase = n0 + wn * 64 + r;
#pragma unroll
  for (int t = 0; t < 4; ++t) {
    const int tc = cbase + t * 16;
#pragma unroll
    for (int mi = 0; mi < 2; ++mi) {
      const int tr = rbase + mi * 16;
      const v8f a = acc[t * 2 + mi];
#pragma unroll
      for (int v = 0; v < 8; ++v) {
        const size_t idx = (size_t)(tr + v) * N + tc;
        float val = a[v];
        if (EP == 0) {
          C[idx] = val;
        } else if (EP == 1) {
          C[idx] += val;
        } else if (EP == 2) {
          float xg = val + bias[tc];
          Cb[idx] = f2bf(0.5f * xg * (1.0f + erff(xg * 0.70710678f)));
        } else {  // EP == 3
          C[idx] += val + bias[tc];
        }
      }
    }
  }
}

// --------------------------- flash attention -------------------------------
// One wave per block: head hh, 16 query rows [qb, qb+15], key chunks of 32,
// online softmax. Q,K: [H][N][DH] bf16, Vt: [H][DH][N] bf16, O: [N][H*DH] bf16
__global__ __launch_bounds__(32)
void attn_kernel(const bf16_t* __restrict__ Q, const bf16_t* __restrict__ Kc,
                 const bf16_t* __restrict__ Vt, bf16_t* __restrict__ O) {
  __shared__ bf16_t Ks[32][72];    // [key][dh]  row-major K chunk
  __shared__ bf16_t Vts[64][40];   // [dh][key]  transposed V chunk
  __shared__ bf16_t Ps[16][40];    // probabilities

  const int lane = threadIdx.x;
  const int hh   = blockIdx.y;
  const int qb   = blockIdx.x * 16;
  const int r    = lane & 15;
  const int koff = (lane < 16) ? 0 : 8;
  const int khl  = (lane < 16) ? 0 : 16;
  const int rowl = (lane < 16) ? 0 : 8;
  const float scale = 0.125f;           // 1/sqrt(64)
  const float NEG = -3.0e38f;

  // Q fragments (rows qb..qb+15, DH split into two K=32 chunks)
  v16bf aq0, aq1;
  {
    const v8bf* q8 = (const v8bf*)(Q + ((size_t)hh * SEQ + (qb + r)) * DHEAD);
    const int g = khl >> 3;
    aq0 = CAT16(q8[g], q8[g + 1]);
    aq1 = CAT16(q8[g + 4], q8[g + 5]);
  }

  float m8[8], l8[8];
  v8f zero = {};
  v8f o[4];
#pragma unroll
  for (int v = 0; v < 8; ++v) { m8[v] = NEG; l8[v] = 0.f; }
#pragma unroll
  for (int t = 0; t < 4; ++t) o[t] = zero;

  const int nkb = qb / 32 + 1;
  for (int kb2 = 0; kb2 < nkb; ++kb2) {
    const int kbase = kb2 * 32;
    {  // stage K chunk (row-major) and V chunk (already transposed in global)
      const int key = kbase + lane;
      const v8bf* kg = (const v8bf*)(Kc + ((size_t)hh * SEQ + key) * DHEAD);
#pragma unroll
      for (int g = 0; g < 8; ++g) *(v8bf*)(&Ks[lane][g * 8]) = kg[g];
#pragma unroll
      for (int rr = 0; rr < 2; ++rr) {
        const int d = lane + rr * 32;
        const v8bf* vg =
            (const v8bf*)(Vt + ((size_t)hh * DHEAD + d) * SEQ + kbase);
#pragma unroll
        for (int g = 0; g < 4; ++g) *(v8bf*)(&Vts[d][g * 8]) = vg[g];
      }
    }
    __syncthreads();

    // S = Q K^T : two 16x16 tiles over 32 keys; B-frag = contiguous dh run
    v8f s0 = zero, s1 = zero;
    {
      v16bf b0 = CAT16(*(const v8bf*)&Ks[r][khl],
                       *(const v8bf*)&Ks[r][khl + 8]);
      v16bf b1 = CAT16(*(const v8bf*)&Ks[r][32 + khl],
                       *(const v8bf*)&Ks[r][32 + khl + 8]);
      s0 = WMMA_BF16(aq0, b0, s0);
      s0 = WMMA_BF16(aq1, b1, s0);
    }
    {
      v16bf b0 = CAT16(*(const v8bf*)&Ks[16 + r][khl],
                       *(const v8bf*)&Ks[16 + r][khl + 8]);
      v16bf b1 = CAT16(*(const v8bf*)&Ks[16 + r][32 + khl],
                       *(const v8bf*)&Ks[16 + r][32 + khl + 8]);
      s1 = WMMA_BF16(aq0, b0, s1);
      s1 = WMMA_BF16(aq1, b1, s1);
    }

    // causal mask + online softmax (row reductions across 16-lane halves)
#pragma unroll
    for (int v = 0; v < 8; ++v) {
      const int row = qb + rowl + v;
      float x0 = s0[v] * scale; if (kbase + r > row)      x0 = NEG;
      float x1 = s1[v] * scale; if (kbase + 16 + r > row) x1 = NEG;
      float mx = fmaxf(x0, x1);
      mx = fmaxf(mx, __shfl_xor(mx, 1, 32));
      mx = fmaxf(mx, __shfl_xor(mx, 2, 32));
      mx = fmaxf(mx, __shfl_xor(mx, 4, 32));
      mx = fmaxf(mx, __shfl_xor(mx, 8, 32));
      const float mnew  = fmaxf(m8[v], mx);
      const float alpha = expf(m8[v] - mnew);
      const float p0 = expf(x0 - mnew);
      const float p1 = expf(x1 - mnew);
      float rs = p0 + p1;
      rs += __shfl_xor(rs, 1, 32);
      rs += __shfl_xor(rs, 2, 32);
      rs += __shfl_xor(rs, 4, 32);
      rs += __shfl_xor(rs, 8, 32);
      l8[v] = l8[v] * alpha + rs;
      m8[v] = mnew;
#pragma unroll
      for (int t = 0; t < 4; ++t) o[t][v] *= alpha;
      Ps[rowl + v][r]      = f2bf(p0);
      Ps[rowl + v][16 + r] = f2bf(p1);
    }
    __syncthreads();

    // O += P * V : A-frag from Ps, B-frag = contiguous key run of Vts
    v16bf ap = CAT16(*(const v8bf*)&Ps[r][khl],
                     *(const v8bf*)&Ps[r][khl + 8]);
#pragma unroll
    for (int t = 0; t < 4; ++t) {
      v16bf bv = CAT16(*(const v8bf*)&Vts[t * 16 + r][khl],
                       *(const v8bf*)&Vts[t * 16 + r][khl + 8]);
      o[t] = WMMA_BF16(ap, bv, o[t]);
    }
    __syncthreads();
  }

  // normalize, store O as [N][H*DH] bf16
#pragma unroll
  for (int v = 0; v < 8; ++v) {
    const float invl = 1.0f / l8[v];
    const int row = qb + rowl + v;
#pragma unroll
    for (int t = 0; t < 4; ++t)
      O[(size_t)row * (NHEAD * DHEAD) + hh * DHEAD + t * 16 + r] =
          f2bf(o[t][v] * invl);
  }
}

// ---------------------------------------------------------------------------
extern "C" void kernel_launch(void* const* d_in, const int* in_sizes, int n_in,
                              void* d_out, int out_size, void* d_ws, size_t ws_size,
                              hipStream_t stream) {
  (void)in_sizes; (void)n_in; (void)out_size; (void)ws_size;

  const int*   x           = (const int*)d_in[0];
  const float* token_emb   = (const float*)d_in[1];
  const float* attn_gamma  = (const float*)d_in[2];
  const float* w_qkv       = (const float*)d_in[3];
  const float* w_attn_out  = (const float*)d_in[4];
  const float* ff_gamma    = (const float*)d_in[5];
  const float* w_ff1       = (const float*)d_in[6];
  const float* b_ff1       = (const float*)d_in[7];
  const float* w_ff2       = (const float*)d_in[8];
  const float* b_ff2       = (const float*)d_in[9];
  const float* final_gamma = (const float*)d_in[10];
  const float* w_logits    = (const float*)d_in[11];
  float* out = (float*)d_out;

  // workspace carve-out (~240 MB total)
  char* ws = (char*)d_ws;
  size_t off = 0;
  auto carve = [&](size_t bytes) -> void* {
    void* p = ws + off;
    off = (off + bytes + 255) & ~(size_t)255;
    return p;
  };
  bf16_t* wqkv_t = (bf16_t*)carve((size_t)DEPTH_L * D_MODEL * QKVDIM * 2);
  bf16_t* wao_t  = (bf16_t*)carve((size_t)DEPTH_L * NHEAD * DHEAD * D_MODEL * 2);
  bf16_t* wff1_t = (bf16_t*)carve((size_t)DEPTH_L * D_MODEL * FFDIM * 2);
  bf16_t* wff2_t = (bf16_t*)carve((size_t)DEPTH_L * FFDIM * D_MODEL * 2);
  bf16_t* wlog_t = (bf16_t*)carve((size_t)D_MODEL * V_SIZE * 2);
  float*  hbuf   = (float*) carve((size_t)SEQ * D_MODEL * 4);
  bf16_t* xnb    = (bf16_t*)carve((size_t)SEQ * D_MODEL * 2);
  float*  qkvbuf = (float*) carve((size_t)SEQ * QKVDIM * 4);
  bf16_t* Qb     = (bf16_t*)carve((size_t)NHEAD * SEQ * DHEAD * 2);
  bf16_t* Kb     = (bf16_t*)carve((size_t)NHEAD * SEQ * DHEAD * 2);
  bf16_t* Vtb    = (bf16_t*)carve((size_t)NHEAD * DHEAD * SEQ * 2);
  bf16_t* aob    = (bf16_t*)carve((size_t)SEQ * D_MODEL * 2);
  bf16_t* ffb    = (bf16_t*)carve((size_t)SEQ * FFDIM * 2);

  // one-time per launch: transpose+convert weights to bf16 [N][K]
  auto tconv = [&](const float* src, bf16_t* dst, int K, int N) {
    transpose_f32_bf16<<<dim3(N / 32, K / 32), dim3(32, 8), 0, stream>>>(
        src, dst, K, N);
  };
  for (int l = 0; l < DEPTH_L; ++l) {
    tconv(w_qkv      + (size_t)l * D_MODEL * QKVDIM,  wqkv_t + (size_t)l * D_MODEL * QKVDIM, D_MODEL, QKVDIM);
    tconv(w_attn_out + (size_t)l * D_MODEL * D_MODEL, wao_t  + (size_t)l * D_MODEL * D_MODEL, D_MODEL, D_MODEL);
    tconv(w_ff1      + (size_t)l * D_MODEL * FFDIM,   wff1_t + (size_t)l * D_MODEL * FFDIM, D_MODEL, FFDIM);
    tconv(w_ff2      + (size_t)l * FFDIM * D_MODEL,   wff2_t + (size_t)l * FFDIM * D_MODEL, FFDIM, D_MODEL);
  }
  tconv(w_logits, wlog_t, D_MODEL, V_SIZE);

  embed_kernel<<<SEQ, 256, 0, stream>>>(x, token_emb, hbuf);

  for (int l = 0; l < DEPTH_L; ++l) {
    // --- attention block ---
    rmsnorm_kernel<<<SEQ, 256, 0, stream>>>(hbuf, attn_gamma + (size_t)l * D_MODEL, xnb);
    gemm_bf16<0><<<dim3(QKVDIM / 128, SEQ / 128), 256, 0, stream>>>(
        xnb, wqkv_t + (size_t)l * D_MODEL * QKVDIM, qkvbuf, nullptr, nullptr,
        SEQ, QKVDIM, D_MODEL);
    rope_split_kernel<<<dim3(SEQ, NHEAD), 64, 0, stream>>>(qkvbuf, Qb, Kb, Vtb);
    attn_kernel<<<dim3(SEQ / 16, NHEAD), 32, 0, stream>>>(Qb, Kb, Vtb, aob);
    gemm_bf16<1><<<dim3(D_MODEL / 128, SEQ / 128), 256, 0, stream>>>(
        aob, wao_t + (size_t)l * D_MODEL * D_MODEL, hbuf, nullptr, nullptr,
        SEQ, D_MODEL, NHEAD * DHEAD);
    // --- feed-forward block ---
    rmsnorm_kernel<<<SEQ, 256, 0, stream>>>(hbuf, ff_gamma + (size_t)l * D_MODEL, xnb);
    gemm_bf16<2><<<dim3(FFDIM / 128, SEQ / 128), 256, 0, stream>>>(
        xnb, wff1_t + (size_t)l * D_MODEL * FFDIM, nullptr, ffb,
        b_ff1 + (size_t)l * FFDIM, SEQ, FFDIM, D_MODEL);
    gemm_bf16<3><<<dim3(D_MODEL / 128, SEQ / 128), 256, 0, stream>>>(
        ffb, wff2_t + (size_t)l * FFDIM * D_MODEL, hbuf, nullptr,
        b_ff2 + (size_t)l * D_MODEL, SEQ, D_MODEL, FFDIM);
  }

  // --- final norm + logits ---
  rmsnorm_kernel<<<SEQ, 256, 0, stream>>>(hbuf, final_gamma, xnb);
  gemm_bf16<0><<<dim3(V_SIZE / 128, SEQ / 128), 256, 0, stream>>>(
      xnb, wlog_t, out, nullptr, nullptr, SEQ, V_SIZE, D_MODEL);
}
